// NetGIN_19765439496792
// MI455X (gfx1250) — matrix operand order repaired and checked by reading, not compiled, not run on hardware
//
#include <hip/hip_runtime.h>
#include <math.h>

// ---------------------------------------------------------------------------
// NetGIN forward on MI455X (gfx1250, wave32, WMMA).
// GEMMs: bf16 WMMA, fp32 accumulate; weights pre-converted to transposed bf16
// (L2/L0-hot, shared by all blocks; KPAD templated -> immediate tile offsets).
// Per K-chunk: one 16-load clause for all 8 B tiles, then 8 back-to-back
// WMMAs (single wait per chunk). A staged per-wave through LDS (no barrier).
// Sparse aggregation: block-per-edges kernel with wave-uniform (scalar) edge
// indices + native f32 global atomics. Working set (<160MB) is L2-resident.
// ---------------------------------------------------------------------------

typedef __attribute__((ext_vector_type(16))) __bf16 v16bf;
typedef __attribute__((ext_vector_type(8)))  __bf16 v8bf;
typedef __attribute__((ext_vector_type(8)))  float  v8f;

static constexpr int NN   = 50000;   // nodes
static constexpr int EE   = 800000;  // edges per set
static constexpr int FIN  = 97;      // input features
static constexpr int DIM  = 128;     // hidden
static constexpr int GG   = 512;     // graphs

__device__ __forceinline__ void atomAddF(float* p, float v) {
    __hip_atomic_fetch_add(p, v, __ATOMIC_RELAXED, __HIP_MEMORY_SCOPE_AGENT);
}

__device__ __forceinline__ v8bf cvt8(float4 a, float4 b) {
    v8bf r;
    r[0] = (__bf16)a.x; r[1] = (__bf16)a.y; r[2] = (__bf16)a.z; r[3] = (__bf16)a.w;
    r[4] = (__bf16)b.x; r[5] = (__bf16)b.y; r[6] = (__bf16)b.z; r[7] = (__bf16)b.w;
    return r;
}

// ---------------------------------------------------------------------------
__global__ void zero_kernel(float* __restrict__ p, long long n) {
    long long i = (long long)blockIdx.x * blockDim.x + threadIdx.x;
    if (i < n) p[i] = 0.0f;
}

// ---------------------------------------------------------------------------
// Wt[col][Kpad] (bf16, zero-padded) <- W[k][col] (f32, [K,128])
__global__ void convert_w_kernel(const float* __restrict__ W, int K,
                                 __bf16* __restrict__ Wt, int Kpad) {
    int i = blockIdx.x * blockDim.x + threadIdx.x;
    if (i >= 128 * Kpad) return;
    const int col = i / Kpad;
    const int k   = i - col * Kpad;
    const float v = (k < K) ? W[(long long)k * 128 + col] : 0.0f;
    Wt[i] = (__bf16)v;
}

// ---------------------------------------------------------------------------
// agg[n,f] = (1+eps) * h[n,f]
__global__ void agg_init_kernel(const float* __restrict__ h, int ldh,
                                const float* __restrict__ epsp,
                                float* __restrict__ agg, int F, int N) {
    int i = blockIdx.x * blockDim.x + threadIdx.x;
    if (i >= N * F) return;
    int n = i / F, f = i - n * F;
    agg[i] = (1.0f + epsp[0]) * h[(long long)n * ldh + f];
}

// agg[dst,f] += h[src,f].  blockDim = 128 (= feature lane), 4 edges per block:
// edge indices are wave-uniform -> scalar loads; lane adds only its f offset.
__global__ __launch_bounds__(128)
void agg_edge_kernel(const float* __restrict__ h, int ldh,
                     const int* __restrict__ ei, int E,
                     float* __restrict__ agg, int F) {
    const int f  = threadIdx.x;
    const int e0 = blockIdx.x * 4;
    if (f >= F) return;
#pragma unroll
    for (int i = 0; i < 4; ++i) {
        const int e = e0 + i;           // E % 4 == 0
        const int s = ei[e];            // uniform across wave -> s_load
        const int d = ei[E + e];
        atomAddF(&agg[(long long)d * F + f], h[(long long)s * ldh + f]);
    }
}

// ---------------------------------------------------------------------------
// C[M,128] = act(A[M,K] @ W[K,128] + bias), W pre-transposed bf16 Wt[col][KPAD].
// 256 threads = 8 wave32; wave w owns rows [blk*128+16w,+16) x 128 cols via 8
// WMMA accumulators. A staged per-wave in LDS (no __syncthreads). All 8 B-tile
// fragments loaded in one clause (immediate offsets), then 8 WMMAs back-to-back.
template <int KPAD>
__global__ __launch_bounds__(256)
void gemm_bias_act_wmma(const float* __restrict__ A, int lda, int M, int K,
                        const __bf16* __restrict__ Wt,    // [128][KPAD]
                        const float* __restrict__ bias,   // [128]
                        float* __restrict__ C, int ldc, int do_relu) {
    __shared__ __align__(16) __bf16 As[128 * 40];   // [row][k], padded stride 40

    const int tid  = threadIdx.x;
    const int wave = tid >> 5;
    const int lane = tid & 31;
    const int rowblk = blockIdx.x * 128;
    const bool rows_ok = (rowblk + 128 <= M);
    const bool lda_vec = ((lda & 3) == 0);

    v8f acc[8];
#pragma unroll
    for (int n = 0; n < 8; ++n)
#pragma unroll
        for (int v = 0; v < 8; ++v) acc[n][v] = 0.0f;

    const int lr = lane & 15;            // row-in-tile / col-in-tile
    const int kh = lane >> 4;            // k-half selector

    for (int k0 = 0; k0 < KPAD; k0 += 32) {
        // --- stage this wave's 16 rows x 32 k of A as bf16 in LDS ---
        {
            const int r  = wave * 16 + (lane >> 1);   // row within block
            const int kb = (lane & 1) * 16;           // 16 k's per thread
            const int gr = rowblk + r;
            const float* ap = A + (long long)gr * lda + k0 + kb;
            __bf16* dst = &As[r * 40 + kb];
            if (rows_ok && lda_vec && (k0 + 32 <= K)) {
                const float4 f0 = ((const float4*)ap)[0];
                const float4 f1 = ((const float4*)ap)[1];
                const float4 f2 = ((const float4*)ap)[2];
                const float4 f3 = ((const float4*)ap)[3];
                *(v8bf*)dst       = cvt8(f0, f1);
                *(v8bf*)(dst + 8) = cvt8(f2, f3);
                if (k0 + 64 <= K) __builtin_prefetch(ap + 32, 0, 1);
            } else {
#pragma unroll
                for (int j = 0; j < 16; ++j) {
                    const int k = k0 + kb + j;
                    const float v = (gr < M && k < K) ? ap[j] : 0.0f;
                    dst[j] = (__bf16)v;
                }
            }
        }

        // --- issue all 8 B tiles as one clause (compile-time immediate offsets);
        //     B fragment: lane holds col = n*16+lr, 16 contiguous K ---
        const __bf16* wbase = Wt + (long long)lr * KPAD + k0 + kh * 16;
        v8bf blo[8], bhi[8];
#pragma unroll
        for (int n = 0; n < 8; ++n) {
            blo[n] = *(const v8bf*)(wbase + n * 16 * KPAD);
            bhi[n] = *(const v8bf*)(wbase + n * 16 * KPAD + 8);
        }

        // --- A fragment (overlaps in-flight B loads):
        //     lane<16 -> K0..15, lane>=16 -> K16..31 ---
        v16bf afrag;
        {
            const int rbase = (wave * 16 + lr) * 40;
            const v8bf a0 = *(const v8bf*)&As[rbase + kh * 16];
            const v8bf a1 = *(const v8bf*)&As[rbase + kh * 16 + 8];
#pragma unroll
            for (int i = 0; i < 8; ++i) { afrag[i] = a0[i]; afrag[8 + i] = a1[i]; }
        }

        // --- 8 back-to-back WMMAs ---
#pragma unroll
        for (int n = 0; n < 8; ++n) {
            v16bf bfrag;
#pragma unroll
            for (int i = 0; i < 8; ++i) { bfrag[i] = blo[n][i]; bfrag[8 + i] = bhi[n][i]; }
            acc[n] = __builtin_amdgcn_wmma_f32_16x16x32_bf16(
                false, afrag, false, bfrag, (short)0, acc[n], false, false);
        }
    }

    // --- epilogue: C/D layout lane<16 -> M=v, lane>=16 -> M=v+8; N=lane%16 ---
    const int rh = kh * 8;
#pragma unroll
    for (int n = 0; n < 8; ++n) {
        const int col = n * 16 + lr;
        const float bv = bias[col];
#pragma unroll
        for (int v = 0; v < 8; ++v) {
            const int row = rowblk + wave * 16 + rh + v;
            if (row < M) {
                float o = acc[n][v] + bv;
                if (do_relu) o = fmaxf(o, 0.0f);
                C[(long long)row * ldc + col] = o;
            }
        }
    }
}

// ---------------------------------------------------------------------------
// BatchNorm statistics: per-column sum / sum-of-squares over N rows (128 cols)
__global__ void bn_stats_kernel(const float* __restrict__ x,
                                float* __restrict__ sums,
                                float* __restrict__ sumsq, int N) {
    const int col = threadIdx.x;    // blockDim.x == 128
    float s = 0.0f, s2 = 0.0f;
    for (int r = blockIdx.x; r < N; r += gridDim.x) {
        const float v = x[(long long)r * 128 + col];
        s += v; s2 += v * v;
    }
    atomAddF(&sums[col], s);
    atomAddF(&sumsq[col], s2);
}

__global__ void bn_apply_kernel(const float* __restrict__ x,
                                const float* __restrict__ sums,
                                const float* __restrict__ sumsq,
                                const float* __restrict__ gamma,
                                const float* __restrict__ beta,
                                float* __restrict__ out, int ldo, int N) {
    int i = blockIdx.x * blockDim.x + threadIdx.x;
    if (i >= N * 128) return;
    const int n = i >> 7, c = i & 127;
    const float invN = 1.0f / (float)N;
    const float m = sums[c] * invN;
    const float var = sumsq[c] * invN - m * m;
    const float xs = (x[i] - m) * rsqrtf(var + 1e-5f);
    out[(long long)n * ldo + c] = gamma[c] * xs + beta[c];
}

// ---------------------------------------------------------------------------
// mean pooling by graph
__global__ void pool_scatter_kernel(const float* __restrict__ hcat,
                                    const int* __restrict__ batch,
                                    float* __restrict__ pooled, int N) {
    int i = blockIdx.x * blockDim.x + threadIdx.x;
    if (i >= N * 512) return;
    const int n = i >> 9, f = i & 511;
    atomAddF(&pooled[((long long)batch[n] << 9) + f], hcat[i]);
}

__global__ void count_kernel(const int* __restrict__ batch,
                             float* __restrict__ counts, int N) {
    int i = blockIdx.x * blockDim.x + threadIdx.x;
    if (i < N) atomAddF(&counts[batch[i]], 1.0f);
}

__global__ void pool_div_kernel(float* __restrict__ pooled,
                                const float* __restrict__ counts) {
    int i = blockIdx.x * blockDim.x + threadIdx.x;
    if (i >= GG * 512) return;
    pooled[i] /= fmaxf(counts[i >> 9], 1.0f);
}

// ---------------------------------------------------------------------------
// final scalar head: out[g] = o3[g,:] . W[:,0] + b
__global__ void fc4_kernel(const float* __restrict__ o3,
                           const float* __restrict__ W,
                           const float* __restrict__ b,
                           float* __restrict__ out, int G) {
    int g = blockIdx.x * blockDim.x + threadIdx.x;
    if (g >= G) return;
    float s = b[0];
#pragma unroll 8
    for (int f = 0; f < 128; ++f) s += o3[g * 128 + f] * W[f];
    out[g] = s;
}

// ---------------------------------------------------------------------------
extern "C" void kernel_launch(void* const* d_in, const int* in_sizes, int n_in,
                              void* d_out, int out_size, void* d_ws, size_t ws_size,
                              hipStream_t stream) {
    (void)in_sizes; (void)out_size; (void)ws_size;
    if (n_in < 76) return;

    const float* x     = (const float*)d_in[0];
    const int*   ei1   = (const int*)d_in[1];
    const int*   ei2   = (const int*)d_in[2];
    const int*   batch = (const int*)d_in[3];
    auto P = [&](int i) { return (const float*)d_in[i]; };

    // workspace layout (floats)
    float* ws     = (float*)d_ws;
    float* agg    = ws;                              // N*128
    float* tmp    = agg    + (size_t)NN * 128;       // N*128
    float* hc     = tmp    + (size_t)NN * 128;       // N*256 (h1|h2 concat)
    float* hcat   = hc     + (size_t)NN * 256;       // N*512 (BN outputs, 4 slices)
    float* pooled = hcat   + (size_t)NN * 512;       // G*512
    float* counts = pooled + (size_t)GG * 512;       // G
    float* o1     = counts + GG;                     // G*128
    float* o2     = o1     + (size_t)GG * 128;       // G*128
    float* o3     = o2     + (size_t)GG * 128;       // G*128
    float* stats  = o3     + (size_t)GG * 128;       // 256 (sums|sumsq)
    __bf16* wbf   = (__bf16*)(stats + 256);          // 128*512 bf16 transposed W

    auto cdiv = [](long long a, long long b) { return (unsigned)((a + b - 1) / b); };
    auto gemm = [&](const float* A, int lda, int M, int K,
                    const float* W, const float* b, float* C, int ldc, int relu) {
        const int Kpad = ((K + 31) / 32) * 32;
        convert_w_kernel<<<cdiv(128LL * Kpad, 256), 256, 0, stream>>>(W, K, wbf, Kpad);
        const unsigned grid = cdiv(M, 128);
        switch (Kpad) {
        case 128:
            gemm_bias_act_wmma<128><<<grid, 256, 0, stream>>>(A, lda, M, K, wbf, b, C, ldc, relu);
            break;
        case 256:
            gemm_bias_act_wmma<256><<<grid, 256, 0, stream>>>(A, lda, M, K, wbf, b, C, ldc, relu);
            break;
        default:
            gemm_bias_act_wmma<512><<<grid, 256, 0, stream>>>(A, lda, M, K, wbf, b, C, ldc, relu);
            break;
        }
    };

    for (int L = 0; L < 4; ++L) {
        const float* hin = (L == 0) ? x : (hcat + (L - 1) * 128);
        const int ldh = (L == 0) ? FIN : 512;
        const int F   = (L == 0) ? FIN : DIM;
        const int base = 4 + L * 16;  // conv1{W1,b1,W2,b2,eps} conv2{...} mlp{W1,b1,W2,b2} bn{g,b}

        for (int c = 0; c < 2; ++c) {
            const int cb = base + c * 5;
            const int* ei = (c == 0) ? ei1 : ei2;
            // agg = (1+eps)*h + scatter_add(h[src] -> dst)
            agg_init_kernel<<<cdiv((long long)NN * F, 256), 256, 0, stream>>>(
                hin, ldh, P(cb + 4), agg, F, NN);
            agg_edge_kernel<<<EE / 4, 128, 0, stream>>>(hin, ldh, ei, EE, agg, F);
            // conv MLP: tmp = relu(agg@W1+b1); hc[:,c*128:] = relu(tmp@W2+b2)
            gemm(agg, F, NN, F, P(cb + 0), P(cb + 1), tmp, 128, 1);
            gemm(tmp, 128, NN, 128, P(cb + 2), P(cb + 3), hc + c * 128, 256, 1);
        }
        // combine MLP: tmp = relu(hc@W1+b1) [256->128]; agg = tmp@W2+b2 (pre-BN)
        gemm(hc, 256, NN, 256, P(base + 10), P(base + 11), tmp, 128, 1);
        gemm(tmp, 128, NN, 128, P(base + 12), P(base + 13), agg, 128, 0);
        // BatchNorm (training-mode batch stats) -> hcat slice L
        zero_kernel<<<1, 256, 0, stream>>>(stats, 256);
        bn_stats_kernel<<<512, 128, 0, stream>>>(agg, stats, stats + 128, NN);
        bn_apply_kernel<<<cdiv((long long)NN * 128, 256), 256, 0, stream>>>(
            agg, stats, stats + 128, P(base + 14), P(base + 15), hcat + L * 128, 512, NN);
    }

    // mean pooling over graphs
    zero_kernel<<<cdiv((long long)GG * 512 + GG, 256), 256, 0, stream>>>(
        pooled, (long long)GG * 512 + GG);
    pool_scatter_kernel<<<cdiv((long long)NN * 512, 256), 256, 0, stream>>>(
        hcat, batch, pooled, NN);
    count_kernel<<<cdiv(NN, 256), 256, 0, stream>>>(batch, counts, NN);
    pool_div_kernel<<<cdiv((long long)GG * 512, 256), 256, 0, stream>>>(pooled, counts);

    // FC head
    gemm(pooled, 512, GG, 512, P(68), P(69), o1, 128, 1);
    gemm(o1, 128, GG, 128, P(70), P(71), o2, 128, 1);
    gemm(o2, 128, GG, 128, P(72), P(73), o3, 128, 1);
    fc4_kernel<<<cdiv(GG, 256), 256, 0, stream>>>(o3, P(74), P(75), (float*)d_out, GG);
}
